// CRF_v2_86595130622090
// MI455X (gfx1250) — matrix operand order repaired and checked by reading, not compiled, not run on hardware
//
#include <hip/hip_runtime.h>

// CRF negative log-likelihood for MI455X (gfx1250, wave32).
// Forward scan: logsumexp recast as exp-space GEMM on v_wmma_f32_16x16x32_f16.
//   E = exp(transitions) preloaded once into WMMA A-layout registers.
//   partition kept in registers (C/D layout: VGPR=tag-row, lane=batch-col).
//   per-step lane<->reg transpose of exp(part-m) via small padded LDS tile.
//   raw v_exp_f32/v_log_f32 (range-safe by construction) + branch-free blend.

typedef _Float16 v16h __attribute__((ext_vector_type(16)));
typedef _Float16 v8h  __attribute__((ext_vector_type(8)));
typedef float    v8f  __attribute__((ext_vector_type(8)));
typedef float    v4f  __attribute__((ext_vector_type(4)));

#define B_        256
#define T_        1024
#define K_        64
#define START_TAG 62
#define STOP_TAG  63
#define LROW      24   // LDS row stride in f16 (48B: 16B-aligned, conflict-light)

#define L2E 1.4426950408889634f   // log2(e)
#define LN2 0.6931471805599453f   // ln(2)

// raw-instruction transcendentals: v_exp_f32 / v_log_f32 (base-2 HW ops)
__device__ __forceinline__ float fast_exp(float x) {
  return __builtin_amdgcn_exp2f(x * L2E);
}
__device__ __forceinline__ float fast_log(float x) {
  return __builtin_amdgcn_logf(x) * LN2;
}

// ---------------------------------------------------------------------------
// Forward algorithm: one wave per 16 batch rows, grid = 16 blocks of 32 thr.
// ---------------------------------------------------------------------------
__launch_bounds__(32)
__global__ void crf_forward_kernel(const float* __restrict__ feats,
                                   const float* __restrict__ trans,
                                   const unsigned char* __restrict__ mask,
                                   float* __restrict__ logz_part) {
  __shared__ __align__(16) _Float16 sh[K_ * LROW];

  const int lane = threadIdx.x;     // 0..31
  const int half = lane >> 4;       // 0 or 1
  const int bl   = lane & 15;       // local batch col 0..15
  const int b    = blockIdx.x * 16 + bl;
  const size_t fbase = (size_t)b * T_ * K_;

  // ---- preload E = exp(transitions) in WMMA A-layout (16 x 32 f16 chunks) --
  // A layout (16-bit): lane = M row (dup in both halves); per half:
  //   elems 0..7  -> K = e + 8*half ; elems 8..15 -> K = 16 + (e-8) + 8*half
  v16h Ea[4][2];
#pragma unroll
  for (int i = 0; i < 4; ++i) {
#pragma unroll
    for (int k = 0; k < 2; ++k) {
#pragma unroll
      for (int e = 0; e < 16; ++e) {
        const int fl   = e + 8 * half + ((e >= 8) ? 8 : 0);
        const int to   = 16 * i + bl;
        const int from = 32 * k + fl;
        Ea[i][k][e] = (_Float16)fast_exp(trans[to * K_ + from]);
      }
    }
  }

  // ---- initial partition: part0[b,to] = feat[b,0,to] + trans[to, START] ---
  // C/D layout: tile j, VGPR r, to = 16j + r + 8*half, batch = bl
  v8f part[4];
#pragma unroll
  for (int j = 0; j < 4; ++j) {
    const float* p = feats + fbase + 16 * j + 8 * half;
    v4f a = *(const v4f*)p;
    v4f c = *(const v4f*)(p + 4);
#pragma unroll
    for (int r = 0; r < 8; ++r) {
      const float fv = (r < 4) ? a[r] : c[r - 4];
      const int   to = 16 * j + r + 8 * half;
      part[j][r] = fv + trans[to * K_ + START_TAG];
    }
  }

  // ---- sequential scan over time ------------------------------------------
  for (int t = 1; t < T_; ++t) {
    // per-batch max over all 64 tags: every reg in a lane shares the batch,
    // lanes n and n+16 hold complementary tag subsets of the same batch.
    float mx = part[0][0];
#pragma unroll
    for (int j = 0; j < 4; ++j)
#pragma unroll
      for (int r = 0; r < 8; ++r) mx = fmaxf(mx, part[j][r]);
    mx = fmaxf(mx, __shfl_xor(mx, 16, 32));
    const float nmx = -mx * L2E;

    // P = exp(part - mx) -> LDS as f16, rows = from-tag, cols = batch
#pragma unroll
    for (int j = 0; j < 4; ++j)
#pragma unroll
      for (int r = 0; r < 8; ++r) {
        const int   from = 16 * j + r + 8 * half;
        const float p    = __builtin_amdgcn_exp2f(fmaf(part[j][r], L2E, nmx));
        sh[from * LROW + bl] = (_Float16)p;
      }

    // B operand (32 x 16): lane = batch col, elems = from rows (table K order)
    v16h Bop[2];
#pragma unroll
    for (int k = 0; k < 2; ++k) {
#pragma unroll
      for (int e = 0; e < 16; ++e) {
        const int fl = e + 8 * half + ((e >= 8) ? 8 : 0);
        Bop[k][e] = sh[(32 * k + fl) * LROW + bl];
      }
    }

    const unsigned char mb  = mask[(size_t)b * T_ + t];
    const unsigned      sel = (unsigned)(-(int)mb);   // all-ones if masked-in
    const float* fb = feats + fbase + (size_t)t * K_;

#pragma unroll
    for (int i = 0; i < 4; ++i) {
      // prefetch next timestep's emission chunk (speculative: safe at t=T-1)
      __builtin_prefetch(fb + K_ + 16 * i + 8 * half, 0, 3);

      v8f acc = {0.f, 0.f, 0.f, 0.f, 0.f, 0.f, 0.f, 0.f};
      acc = __builtin_amdgcn_wmma_f32_16x16x32_f16(
          false, Ea[i][0], false, Bop[0], (short)0, acc, false, false);
      acc = __builtin_amdgcn_wmma_f32_16x16x32_f16(
          false, Ea[i][1], false, Bop[1], (short)0, acc, false, false);

      const float* p = fb + 16 * i + 8 * half;
      v4f ea = *(const v4f*)p;
      v4f eb = *(const v4f*)(p + 4);
#pragma unroll
      for (int r = 0; r < 8; ++r) {
        const float em = (r < 4) ? ea[r] : eb[r - 4];
        // nv = em + mx + ln2 * log2(acc)   (acc >= exp(min trans) > 0: safe)
        const float lg = __builtin_amdgcn_logf(acc[r]);
        const float nv = fmaf(LN2, lg, em + mx);
        // branch-free masked update (bit blend -> v_bfi/v_cndmask, no EXEC)
        const unsigned pb = (__float_as_uint(nv) & sel) |
                            (__float_as_uint(part[i][r]) & ~sel);
        part[i][r] = __uint_as_float(pb);
      }
    }
  }

  // ---- transition into STOP + per-batch logsumexp -------------------------
  float vv[4][8];
  float m2 = -1e30f;
#pragma unroll
  for (int j = 0; j < 4; ++j)
#pragma unroll
    for (int r = 0; r < 8; ++r) {
      const int to = 16 * j + r + 8 * half;
      vv[j][r] = part[j][r] + trans[STOP_TAG * K_ + to];
      m2 = fmaxf(m2, vv[j][r]);
    }
  m2 = fmaxf(m2, __shfl_xor(m2, 16, 32));

  float s = 0.f;
#pragma unroll
  for (int j = 0; j < 4; ++j)
#pragma unroll
    for (int r = 0; r < 8; ++r) s += fast_exp(vv[j][r] - m2);
  s += __shfl_xor(s, 16, 32);

  float lz = m2 + fast_log(s);     // per-batch logZ contribution
  lz = (half == 0) ? lz : 0.f;     // lanes 16..31 duplicate -> zero
#pragma unroll
  for (int o = 16; o >= 1; o >>= 1) lz += __shfl_xor(lz, o, 32);
  if (lane == 0) logz_part[blockIdx.x] = lz;
}

// ---------------------------------------------------------------------------
// Gold path score: one block per batch row, deterministic tree reduction.
// ---------------------------------------------------------------------------
__launch_bounds__(256)
__global__ void crf_gold_kernel(const float* __restrict__ feats,
                                const float* __restrict__ trans,
                                const int* __restrict__ y,
                                const unsigned char* __restrict__ mask,
                                float* __restrict__ gold_part) {
  const int b = blockIdx.x;
  __shared__ float red[256];
  float acc = 0.f;
  for (int t = threadIdx.x; t < T_; t += 256) {
    const size_t bt = (size_t)b * T_ + t;
    if (mask[bt]) {
      const int yt   = y[bt];
      const int prev = (t == 0) ? START_TAG : y[bt - 1];
      acc += feats[bt * K_ + yt] + trans[yt * K_ + prev];
      const bool is_last = (t == T_ - 1) || (mask[bt + 1] == 0);
      if (is_last) acc += trans[STOP_TAG * K_ + yt];   // end_energy
    }
  }
  red[threadIdx.x] = acc;
  __syncthreads();
  for (int s = 128; s > 0; s >>= 1) {
    if (threadIdx.x < s) red[threadIdx.x] += red[threadIdx.x + s];
    __syncthreads();
  }
  if (threadIdx.x == 0) gold_part[b] = red[0];
}

// ---------------------------------------------------------------------------
// Final combine: out = logZ - gold_score (tiny, deterministic).
// ---------------------------------------------------------------------------
__global__ void crf_final_kernel(const float* __restrict__ logz_part,
                                 const float* __restrict__ gold_part,
                                 float* __restrict__ out) {
  if (threadIdx.x == 0 && blockIdx.x == 0) {
    float lz = 0.f, g = 0.f;
    for (int i = 0; i < 16; ++i) lz += logz_part[i];
    for (int i = 0; i < B_; ++i) g += gold_part[i];
    out[0] = lz - g;
  }
}

// ---------------------------------------------------------------------------
extern "C" void kernel_launch(void* const* d_in, const int* in_sizes, int n_in,
                              void* d_out, int out_size, void* d_ws, size_t ws_size,
                              hipStream_t stream) {
  const float*         feats = (const float*)d_in[0];          // (B,T,K) f32
  const float*         trans = (const float*)d_in[1];          // (K,K)   f32
  const int*           y     = (const int*)d_in[2];            // (B,T)   i32
  const unsigned char* mask  = (const unsigned char*)d_in[3];  // (B,T)   bool

  float* ws        = (float*)d_ws;
  float* logz_part = ws;        // 16 floats
  float* gold_part = ws + 16;   // 256 floats

  crf_forward_kernel<<<B_ / 16, 32, 0, stream>>>(feats, trans, mask, logz_part);
  crf_gold_kernel<<<B_, 256, 0, stream>>>(feats, trans, y, mask, gold_part);
  crf_final_kernel<<<1, 32, 0, stream>>>(logz_part, gold_part, (float*)d_out);
}
